// GQAttention_979252543862
// MI455X (gfx1250) — matrix operand order repaired and checked by reading, not compiled, not run on hardware
//
#include <hip/hip_runtime.h>
#include <hip/hip_bf16.h>
#include <math.h>

// Problem constants (match reference)
#define BDIM 2
#define SEQ  2048
#define HIDN 2048
#define NH   16
#define NKV  4
#define HD   128
#define EPSV 1e-5f

typedef __bf16 bf16_t;
typedef __attribute__((ext_vector_type(16))) __bf16 v16bf;
typedef __attribute__((ext_vector_type(8)))  __bf16 v8bf;
typedef __attribute__((ext_vector_type(8)))  float  v8f;

__device__ __forceinline__ int lane_id() { return (int)(threadIdx.x & 31u); }

// D = A(16x32 bf16) * B(32x16 bf16) + C(16x16 f32)
__device__ __forceinline__ v8f wmma_bf16(v16bf a, v16bf b, v8f c) {
  return __builtin_amdgcn_wmma_f32_16x16x32_bf16(false, a, false, b, (short)0, c,
                                                 false, false);
}

// A-fragment (16x32, row-major source, leading dim ld).
// Lane l<16: row=l, K = 0..15 contiguous; lane>=16: row=l-16, K = 16..31.
__device__ __forceinline__ v16bf load_a_frag(const bf16_t* p, int ld) {
  int l = lane_id();
  int row = l & 15, half = l >> 4;
  const bf16_t* q = p + (size_t)row * ld + half * 16;
  return *(const v16bf*)q;
}

// B-fragment where B[k][n] = Wrow[n][k] (row-major W, rows are the N index).
// Lane l<16: n=l, K=0..15 contiguous (32B load); lane>=16: n=l-16, K=16..31.
__device__ __forceinline__ v16bf load_b_frag_t(const bf16_t* p, size_t ld) {
  int l = lane_id();
  int n = l & 15, half = l >> 4;
  return *(const v16bf*)(p + (size_t)n * ld + half * 16);
}

// Per-wave LDS ordering (waves in a block are divergent -> no s_barrier).
__device__ __forceinline__ void lds_fence_wave() {
  __builtin_amdgcn_wave_barrier();
  asm volatile("s_wait_dscnt 0" ::: "memory");
  __builtin_amdgcn_wave_barrier();
}

__device__ __forceinline__ unsigned pack_bf16(float a, float b) {
  bf16_t x = (bf16_t)a, y = (bf16_t)b;
  unsigned short xb = __builtin_bit_cast(unsigned short, x);
  unsigned short yb = __builtin_bit_cast(unsigned short, y);
  return ((unsigned)yb << 16) | (unsigned)xb;
}

__global__ void cvt_f32_bf16(const float* __restrict__ in, bf16_t* __restrict__ out, int n) {
  int i = blockIdx.x * blockDim.x + threadIdx.x;
  int stride = gridDim.x * blockDim.x;
  for (; i < n; i += stride) out[i] = (bf16_t)in[i];
}

// One block = 16 sequence rows x 128 (one head). 8 waves, one 16x16 WMMA tile each,
// K-loop over HIDN. Then fused per-row RMSNorm + RoPE (Q/K) via LDS.
// NORMROPE=true  : out layout [B, heads, S, D] (row-major per head)   -- Q, K
// NORMROPE=false : out layout [B, heads, D, S] (TRANSPOSED)           -- V
template <bool NORMROPE>
__global__ void proj_kernel(const bf16_t* __restrict__ X, const bf16_t* __restrict__ W,
                            const float* __restrict__ normw,
                            const float* __restrict__ cosT, const float* __restrict__ sinT,
                            bf16_t* __restrict__ out, int nHeads) {
  __shared__ float tile[16][HD + 4];
  int row0 = blockIdx.x * 16;
  int head = blockIdx.y;
  int b = row0 / SEQ, s0 = row0 % SEQ;
  int wave = threadIdx.x >> 5;
  int n0 = wave * 16;

  v8f acc = {};
  const bf16_t* xrow = X + (size_t)row0 * HIDN;
  const bf16_t* wrow = W + (size_t)(head * HD + n0) * HIDN;
  for (int k0 = 0; k0 < HIDN; k0 += 32) {
    v16bf a  = load_a_frag(xrow + k0, HIDN);
    v16bf bb = load_b_frag_t(wrow + k0, HIDN);
    acc = wmma_bf16(a, bb, acc);
  }

  int l = lane_id(), n = l & 15, half = l >> 4;
#pragma unroll
  for (int r = 0; r < 8; ++r) tile[r + 8 * half][n0 + n] = acc[r];
  __syncthreads();

  int row = threadIdx.x >> 4, c16 = threadIdx.x & 15;
  int s = s0 + row;
  if (NORMROPE) {
    // 16 threads per row; they are 16 contiguous lanes of one wave half.
    float ss = 0.f;
#pragma unroll
    for (int j = 0; j < 8; ++j) { float v = tile[row][c16 + 16 * j]; ss += v * v; }
    ss += __shfl_xor(ss, 1, 32); ss += __shfl_xor(ss, 2, 32);
    ss += __shfl_xor(ss, 4, 32); ss += __shfl_xor(ss, 8, 32);
    float rstd = rsqrtf(ss * (1.0f / (float)HD) + EPSV);

    bf16_t* orow = out + ((size_t)(b * nHeads + head) * SEQ + s) * HD;
#pragma unroll
    for (int j = 0; j < 8; ++j) {
      int c = c16 + 16 * j;
      float v = tile[row][c] * rstd * normw[c];
      float o;
      if (c < 64) {
        float pr = tile[row][c + 64] * rstd * normw[c + 64];
        o = v * cosT[s * 64 + c] - pr * sinT[s * 64 + c];
      } else {
        int c2 = c - 64;
        float pr = tile[row][c2] * rstd * normw[c2];
        o = v * cosT[s * 64 + c2] + pr * sinT[s * 64 + c2];
      }
      orow[c] = (bf16_t)o;
    }
  } else {
    // V: transposed store [B, heads, D, S] so attention's P@V B-fragments are
    // contiguous 32B loads along the key dimension.
    bf16_t* obase = out + ((size_t)(b * nHeads + head) * HD) * SEQ + s;
#pragma unroll
    for (int j = 0; j < 8; ++j) {
      int c = c16 + 16 * j;
      obase[(size_t)c * SEQ] = (bf16_t)tile[row][c];
    }
  }
}

// Flash attention: one wave per (b, h, 16-row query tile), 32-key tiles.
// Online softmax per-row via LDS (scalar m/l state, no lane shuffles).
// V is pre-transposed [B, KV, D, S]. AO layout [B, S, H*D].
__global__ void __launch_bounds__(256, 1)
attn_kernel(const bf16_t* __restrict__ Q, const bf16_t* __restrict__ K,
            const bf16_t* __restrict__ Vt, bf16_t* __restrict__ AO) {
  __shared__ __align__(16) float  sbuf[8][16][36];   // scaled+masked scores
  __shared__ __align__(32) bf16_t pbuf[8][16][32];   // P tile (A-fragment source)
  __shared__ __align__(16) float  abuf[8][16];       // alpha broadcast
  __shared__ __align__(16) float  lbuf[8][16];       // 1/l broadcast

  int wv = threadIdx.x >> 5;
  int g  = blockIdx.x * 8 + wv;
  int qt = g & 127;            // S/16 = 128 query tiles
  int h  = (g >> 7) & 15;
  int b  = g >> 11;
  int q0 = qt * 16;
  int kvh = h >> 2;            // GQA group of 4
  int l = lane_id(), n = l & 15, half = l >> 4;

  const bf16_t* qbase = Q + ((size_t)(b * NH + h) * SEQ + q0) * HD;
  v16bf qa[4];
#pragma unroll
  for (int kc = 0; kc < 4; ++kc) qa[kc] = load_a_frag(qbase + kc * 32, HD);

  v8f o[8];
#pragma unroll
  for (int nt = 0; nt < 8; ++nt) o[nt] = v8f{};
  float m_row = -__builtin_inff();  // valid in lanes 0..15 (row = lane)
  float l_row = 0.f;

  const float scale = 0.08838834764831845f;  // 1/sqrt(128)
  const float NEGINF = -__builtin_inff();
  int ntiles = (q0 + 47) / 32;  // ceil((q0+16)/32); k0 <= q0 always

  const bf16_t* kvbase = K + ((size_t)(b * NKV + kvh) * SEQ) * HD;
  const bf16_t* vtb    = Vt + ((size_t)(b * NKV + kvh) * HD) * SEQ;

  for (int kt = 0; kt < ntiles; ++kt) {
    int k0 = kt * 32;
    const bf16_t* kbase = kvbase + (size_t)k0 * HD;

    // Warm L2/WGP$ for the next key tile while this one computes.
    if (kt + 1 < ntiles) {
      __builtin_prefetch(kbase + (size_t)32 * HD + (size_t)n * HD, 0, 1);
      __builtin_prefetch(vtb + (size_t)(l * 4) * SEQ + k0 + 32, 0, 1);
    }

    // ---- scores: issue all 8 K B-fragments, then two 4-WMMA chains ----
    v16bf kb0[4], kb1[4];
#pragma unroll
    for (int kc = 0; kc < 4; ++kc) kb0[kc] = load_b_frag_t(kbase + kc * 32, HD);
#pragma unroll
    for (int kc = 0; kc < 4; ++kc)
      kb1[kc] = load_b_frag_t(kbase + (size_t)16 * HD + kc * 32, HD);

    v8f sc0 = {}, sc1 = {};
#pragma unroll
    for (int kc = 0; kc < 4; ++kc) sc0 = wmma_bf16(qa[kc], kb0[kc], sc0);
#pragma unroll
    for (int kc = 0; kc < 4; ++kc) sc1 = wmma_bf16(qa[kc], kb1[kc], sc1);

#pragma unroll
    for (int st = 0; st < 2; ++st) {
      v8f sc = st ? sc1 : sc0;
      int key = k0 + st * 16 + n;
#pragma unroll
      for (int r = 0; r < 8; ++r) {
        int qrow = q0 + r + 8 * half;
        float v = sc[r] * scale;
        if (key > qrow) v = NEGINF;  // causal mask
        sbuf[wv][r + 8 * half][st * 16 + n] = v;
      }
    }
    lds_fence_wave();

    // ---- per-row online softmax: lanes 0..15 own one row each ----
    if (l < 16) {
      const float4* srow4 = (const float4*)&sbuf[wv][l][0];
      float mx = NEGINF;
#pragma unroll
      for (int j = 0; j < 8; ++j) {
        float4 t = srow4[j];
        mx = fmaxf(mx, fmaxf(fmaxf(t.x, t.y), fmaxf(t.z, t.w)));
      }
      float m_new = fmaxf(m_row, mx);
      float alpha = __expf(m_row - m_new);
      m_row = m_new;
      float rs = 0.f;
      uint4* prow4 = (uint4*)&pbuf[wv][l][0];
#pragma unroll
      for (int j = 0; j < 4; ++j) {
        float4 t0 = srow4[2 * j], t1 = srow4[2 * j + 1];
        float p0 = __expf(t0.x - m_new), p1 = __expf(t0.y - m_new);
        float p2 = __expf(t0.z - m_new), p3 = __expf(t0.w - m_new);
        float p4 = __expf(t1.x - m_new), p5 = __expf(t1.y - m_new);
        float p6 = __expf(t1.z - m_new), p7 = __expf(t1.w - m_new);
        rs += ((p0 + p1) + (p2 + p3)) + ((p4 + p5) + (p6 + p7));
        uint4 w;
        w.x = pack_bf16(p0, p1); w.y = pack_bf16(p2, p3);
        w.z = pack_bf16(p4, p5); w.w = pack_bf16(p6, p7);
        prow4[j] = w;
      }
      l_row = l_row * alpha + rs;
      abuf[wv][l] = alpha;
    }
    lds_fence_wave();

    // ---- rescale accumulators by alpha (broadcast via LDS) ----
    const float4* ab = (const float4*)&abuf[wv][8 * half];
    float4 a0 = ab[0], a1 = ab[1];
    float al[8] = {a0.x, a0.y, a0.z, a0.w, a1.x, a1.y, a1.z, a1.w};
#pragma unroll
    for (int nt = 0; nt < 8; ++nt)
#pragma unroll
      for (int r = 0; r < 8; ++r) o[nt][r] *= al[r];

    // ---- P @ V: full 16x32 A-fragment, 8 WMMAs; V transposed -> 32B B loads ----
    v16bf pa = load_a_frag(&pbuf[wv][0][0], 32);
    const bf16_t* vtbase = vtb + k0;
#pragma unroll
    for (int nt = 0; nt < 8; ++nt) {
      v16bf vb = load_b_frag_t(vtbase + (size_t)(nt * 16) * SEQ, SEQ);
      o[nt] = wmma_bf16(pa, vb, o[nt]);
    }
    lds_fence_wave();  // LDS reads done before next iteration overwrites sbuf/pbuf
  }

  // ---- final 1/l broadcast and store ----
  if (l < 16) lbuf[wv][l] = 1.0f / l_row;
  lds_fence_wave();
  const float4* ib = (const float4*)&lbuf[wv][8 * half];
  float4 i0 = ib[0], i1 = ib[1];
  float inv[8] = {i0.x, i0.y, i0.z, i0.w, i1.x, i1.y, i1.z, i1.w};

  bf16_t* ao = AO + ((size_t)b * SEQ + q0) * (NH * HD) + h * HD;
#pragma unroll
  for (int nt = 0; nt < 8; ++nt)
#pragma unroll
    for (int r = 0; r < 8; ++r)
      ao[(size_t)(r + 8 * half) * (NH * HD) + nt * 16 + n] = (bf16_t)(o[nt][r] * inv[r]);
}

// out[m, o] = sum_k AO[m, k] * WO[o, k]; f32 output.
__global__ void oproj_kernel(const bf16_t* __restrict__ AO, const bf16_t* __restrict__ WO,
                             float* __restrict__ out) {
  int row0 = blockIdx.x * 16;
  int wave = threadIdx.x >> 5;
  int n0 = blockIdx.y * 128 + wave * 16;
  v8f acc = {};
  const bf16_t* arow = AO + (size_t)row0 * (NH * HD);
  const bf16_t* wrow = WO + (size_t)n0 * (NH * HD);
  for (int k0 = 0; k0 < NH * HD; k0 += 32) {
    v16bf a  = load_a_frag(arow + k0, NH * HD);
    v16bf bb = load_b_frag_t(wrow + k0, NH * HD);
    acc = wmma_bf16(a, bb, acc);
  }
  int l = lane_id(), n = l & 15, half = l >> 4;
#pragma unroll
  for (int r = 0; r < 8; ++r)
    out[(size_t)(row0 + r + 8 * half) * HIDN + n0 + n] = acc[r];
}

extern "C" void kernel_launch(void* const* d_in, const int* in_sizes, int n_in,
                              void* d_out, int out_size, void* d_ws, size_t ws_size,
                              hipStream_t stream) {
  (void)in_sizes; (void)n_in; (void)out_size; (void)ws_size;
  const float* x   = (const float*)d_in[0];
  const float* rc  = (const float*)d_in[1];
  const float* rs  = (const float*)d_in[2];
  const float* wq  = (const float*)d_in[3];
  const float* wk  = (const float*)d_in[4];
  const float* wv  = (const float*)d_in[5];
  const float* wo  = (const float*)d_in[6];
  const float* qnw = (const float*)d_in[7];
  const float* knw = (const float*)d_in[8];
  float* out = (float*)d_out;

  char* ws = (char*)d_ws;
  size_t off = 0;
  auto carve = [&](size_t elems) {
    bf16_t* p = (bf16_t*)(ws + off);
    off += ((elems * sizeof(bf16_t) + 255) / 256) * 256;
    return p;
  };
  const size_t nx  = (size_t)BDIM * SEQ * HIDN;
  const size_t nwq = (size_t)NH * HD * HIDN;
  const size_t nwk = (size_t)NKV * HD * HIDN;
  bf16_t* x_b  = carve(nx);
  bf16_t* wq_b = carve(nwq);
  bf16_t* wk_b = carve(nwk);
  bf16_t* wv_b = carve(nwk);
  bf16_t* wo_b = carve(nwq);
  bf16_t* q_b  = carve((size_t)BDIM * NH  * SEQ * HD);
  bf16_t* k_b  = carve((size_t)BDIM * NKV * SEQ * HD);
  bf16_t* vt_b = carve((size_t)BDIM * NKV * HD * SEQ);   // transposed V
  bf16_t* ao_b = carve((size_t)BDIM * SEQ * NH * HD);

  cvt_f32_bf16<<<4096, 256, 0, stream>>>(x,  x_b,  (int)nx);
  cvt_f32_bf16<<<4096, 256, 0, stream>>>(wq, wq_b, (int)nwq);
  cvt_f32_bf16<<<2048, 256, 0, stream>>>(wk, wk_b, (int)nwk);
  cvt_f32_bf16<<<2048, 256, 0, stream>>>(wv, wv_b, (int)nwk);
  cvt_f32_bf16<<<4096, 256, 0, stream>>>(wo, wo_b, (int)nwq);

  dim3 gq(BDIM * SEQ / 16, NH);
  dim3 gkv(BDIM * SEQ / 16, NKV);
  proj_kernel<true ><<<gq,  256, 0, stream>>>(x_b, wq_b, qnw, rc, rs, q_b, NH);
  proj_kernel<true ><<<gkv, 256, 0, stream>>>(x_b, wk_b, knw, rc, rs, k_b, NKV);
  proj_kernel<false><<<gkv, 256, 0, stream>>>(x_b, wv_b, nullptr, rc, rs, vt_b, NKV);

  attn_kernel<<<(BDIM * NH * (SEQ / 16)) / 8, 256, 0, stream>>>(q_b, k_b, vt_b, ao_b);

  dim3 go(BDIM * SEQ / 16, HIDN / 128);
  oproj_kernel<<<go, 256, 0, stream>>>(ao_b, wo_b, out);
}
